// InterpolatedCharacterEmbed_300647711242
// MI455X (gfx1250) — matrix operand order repaired and checked by reading, not compiled, not run on hardware
//
#include <hip/hip_runtime.h>
#include <hip/hip_bf16.h>
#include <stdint.h>

// Problem constants (from reference setup_inputs)
constexpr int BB = 16;     // batch
constexpr int SS = 4096;   // audio length (S)
constexpr int TT = 1024;   // text length (T)
constexpr int DD = 512;    // model dim (D)

typedef _Float16 v16h __attribute__((ext_vector_type(16)));
typedef _Float16 v4h  __attribute__((ext_vector_type(4)));
typedef float    v8f  __attribute__((ext_vector_type(8)));
typedef float    v4f  __attribute__((ext_vector_type(4)));
typedef uint32_t v4u  __attribute__((ext_vector_type(4)));

union FragU {
    v16h v;
    v4u  q[2];
};

__device__ __forceinline__ float silu_f(float x) {
    return x * (1.0f / (1.0f + __expf(-x)));
}

// ---------------------------------------------------------------------------
// Kernel 1: per-batch lengths (audio len from mask, text len from text >= 0)
// params[2*b+0] = audio_len (float), params[2*b+1] = text_len (float)
// mask is a jnp bool array -> 1 byte per element (numpy bool layout).
// ---------------------------------------------------------------------------
__global__ void prep_lengths_kernel(const int* __restrict__ text,
                                    const unsigned char* __restrict__ mask,
                                    float* __restrict__ params) {
    __shared__ int red[256];
    const int b = blockIdx.x;
    const int t = threadIdx.x;

    int am = 0;
    for (int i = t; i < SS; i += 256) am += (mask[b * SS + i] != 0) ? 1 : 0;
    int tm = 0;
    for (int i = t; i < TT; i += 256) tm += (text[b * TT + i] >= 0) ? 1 : 0;

    red[t] = am;
    __syncthreads();
    for (int o = 128; o > 0; o >>= 1) {
        if (t < o) red[t] += red[t + o];
        __syncthreads();
    }
    if (t == 0) params[2 * b + 0] = (float)red[0];
    __syncthreads();

    red[t] = tm;
    __syncthreads();
    for (int o = 128; o > 0; o >>= 1) {
        if (t < o) red[t] += red[t + o];
        __syncthreads();
    }
    if (t == 0) params[2 * b + 1] = (float)red[0];
}

// ---------------------------------------------------------------------------
// Kernel 2: w2 [k][n] f32  ->  w2t [n][k] f16  (B operand, L2-resident 512KB)
// ---------------------------------------------------------------------------
__global__ void prep_w2t_kernel(const float* __restrict__ w2,
                                _Float16* __restrict__ w2t) {
    const int tid = blockIdx.x * 256 + threadIdx.x;   // 0 .. 262143
    const int n = tid >> 9;
    const int k = tid & (DD - 1);
    w2t[tid] = (_Float16)w2[(size_t)k * DD + n];
}

// ---------------------------------------------------------------------------
// Kernel 3: fused  silu(pos*w1+b1) @ w2  +  b2  +  embed-interp,  masked.
// Block = 256 threads (8 wave32).  Tile: 32 rows (M) x 128 cols (N).
// Each wave: one 16-col strip, two 16x16 f32 accumulators, 16 K-steps of
// v_wmma_f32_16x16x32_f16.
// ---------------------------------------------------------------------------
constexpr int A_STRIDE = 520;  // halves; 1040B rows: 16B-aligned + bank-spread

__global__ void __launch_bounds__(256)
fused_embed_mlp_kernel(const int* __restrict__ text,
                       const float* __restrict__ embed,
                       const float* __restrict__ w1,
                       const float* __restrict__ b1,
                       const float* __restrict__ b2,
                       const _Float16* __restrict__ w2t,
                       const float* __restrict__ params,
                       float* __restrict__ out) {
    __shared__ _Float16 A_sh[32 * A_STRIDE];   // ~33 KB f16 A panel
    __shared__ int   sTokLo[32];
    __shared__ int   sTokHi[32];
    __shared__ float sW[32];
    __shared__ float sValid[32];

    const int t  = threadIdx.x;
    const int m0 = blockIdx.x * 32;      // global row base (m = b*S + s)
    const int n0 = blockIdx.y * 128;     // global col base
    const int b  = m0 >> 12;             // S = 4096 rows per batch

    const float al = params[2 * b + 0];
    const float tl = params[2 * b + 1];

    // ---- per-row epilogue params (interp indices/weights, mask) ----
    if (t < 32) {
        const int   s     = (m0 + t) & (SS - 1);
        const float valid = ((float)s < al) ? 1.0f : 0.0f;
        float src = ((float)s + 0.5f) * tl / al - 0.5f;
        src = fminf(fmaxf(src, 0.0f), tl - 1.0f);
        const int   lo = (int)floorf(src);
        const int   hi = min(lo + 1, (int)tl - 1);
        const float w  = src - (float)lo;
        int tokLo = text[b * TT + lo];
        int tokHi = text[b * TT + hi];
        if (tokLo < 0) tokLo = 0;
        if (tokHi < 0) tokHi = 0;
        sTokLo[t] = tokLo;
        sTokHi[t] = tokHi;
        sW[t]     = w;
        sValid[t] = valid;
    }

    // ---- stage A panel: h[r][k] = silu(pos_r * w1[k] + b1[k]) as f16 ----
    {
        const int r  = t >> 3;                 // 0..31
        const int k0 = (t & 7) * 64;           // 64 K per thread
        const int s  = (m0 + r) & (SS - 1);
        const float pos = (al > 1.0f) ? (tl * (float)s / fmaxf(al - 1.0f, 1.0f))
                                      : 0.0f;
        for (int k = k0; k < k0 + 64; k += 4) {
            const v4f wv = *(const v4f*)(w1 + k);
            const v4f bv = *(const v4f*)(b1 + k);
            v4h hv;
            hv[0] = (_Float16)silu_f(pos * wv[0] + bv[0]);
            hv[1] = (_Float16)silu_f(pos * wv[1] + bv[1]);
            hv[2] = (_Float16)silu_f(pos * wv[2] + bv[2]);
            hv[3] = (_Float16)silu_f(pos * wv[3] + bv[3]);
            *(v4h*)(&A_sh[r * A_STRIDE + k]) = hv;
        }
    }
    __syncthreads();

    // ---- WMMA GEMM: 2 accumulators per wave, K = 512 in steps of 32 ----
    const int lane  = t & 31;
    const int nw    = t >> 5;                          // wave id = N strip
    const int col   = n0 + nw * 16 + (lane & 15);      // this lane's N column
    const int kOffA = (lane & 16) ? 8 : 0;             // ISA 16-bit A striping
    const int kOffB = (lane & 16) ? 8 : 0;             // ISA 16-bit B striping
    const int rowA  = lane & 15;

    const _Float16* bbase = w2t + (size_t)col * DD + kOffB;
    const _Float16* abase = &A_sh[rowA * A_STRIDE + kOffA];

    v8f acc0 = {};
    v8f acc1 = {};

#pragma unroll 4
    for (int kb = 0; kb < 16; ++kb) {
        FragU bf, a0, a1;
        const _Float16* bp = bbase + kb * 32;
        bf.q[0] = *(const v4u*)(bp);           // K = kOffB + 0..7
        bf.q[1] = *(const v4u*)(bp + 16);      // K = 16 + kOffB + 0..7

        const _Float16* ap = abase + kb * 32;
        a0.q[0] = *(const v4u*)(ap);           // K = kOffA + 0..7
        a0.q[1] = *(const v4u*)(ap + 16);      // K = 16 + kOffA + 0..7
        const _Float16* ap1 = ap + 16 * A_STRIDE;
        a1.q[0] = *(const v4u*)(ap1);
        a1.q[1] = *(const v4u*)(ap1 + 16);

        acc0 = __builtin_amdgcn_wmma_f32_16x16x32_f16(
            false, a0.v, false, bf.v, (short)0, acc0, false, false);
        acc1 = __builtin_amdgcn_wmma_f32_16x16x32_f16(
            false, a1.v, false, bf.v, (short)0, acc1, false, false);
    }

    // ---- fused epilogue: + b2 + embed interpolation, masked store ----
    const float b2n   = b2[col];
    const int   rbase = (lane & 16) ? 8 : 0;   // C/D VGPR->row mapping

#pragma unroll
    for (int sub = 0; sub < 2; ++sub) {
        const v8f acc = sub ? acc1 : acc0;
#pragma unroll
        for (int v = 0; v < 8; ++v) {
            const int   r     = sub * 16 + rbase + v;
            const float w     = sW[r];
            const float valid = sValid[r];
            const float elo   = embed[(size_t)sTokLo[r] * DD + col];
            const float ehi   = embed[(size_t)sTokHi[r] * DD + col];
            const float val   = acc[v] + b2n + (1.0f - w) * elo + w * ehi;
            out[(size_t)(m0 + r) * DD + col] = valid * val;
        }
    }
}

// ---------------------------------------------------------------------------
// Launch. Inputs (setup_inputs order):
//   0: text  [B,T] int32      1: mask [B,S] bool (1 byte)   2: max_seq_len
//   3: embed [V,D] f32        4: w1 [D] f32                 5: b1 [D] f32
//   6: w2 [D,D] f32           7: b2 [D] f32
// Output: [B,S,D] f32.  Workspace: 512KB w2t(f16) + 128B batch params.
// ---------------------------------------------------------------------------
extern "C" void kernel_launch(void* const* d_in, const int* in_sizes, int n_in,
                              void* d_out, int out_size, void* d_ws, size_t ws_size,
                              hipStream_t stream) {
    const int*           text  = (const int*)d_in[0];
    const unsigned char* mask  = (const unsigned char*)d_in[1];
    const float*         embed = (const float*)d_in[3];
    const float*         w1    = (const float*)d_in[4];
    const float*         b1    = (const float*)d_in[5];
    const float*         w2    = (const float*)d_in[6];
    const float*         b2    = (const float*)d_in[7];
    float*               out   = (float*)d_out;

    _Float16* w2t    = (_Float16*)d_ws;                                  // 512 KB
    float*    params = (float*)((char*)d_ws + (size_t)DD * DD * 2);      // 128 B

    prep_lengths_kernel<<<BB, 256, 0, stream>>>(text, mask, params);
    prep_w2t_kernel<<<(DD * DD) / 256, 256, 0, stream>>>(w2, w2t);

    dim3 grid((BB * SS) / 32, DD / 128);   // 2048 x 4
    fused_embed_mlp_kernel<<<grid, 256, 0, stream>>>(
        text, embed, w1, b1, b2, w2t, params, out);
}
